// TopDownSegHead_86320252715180
// MI455X (gfx1250) — compile-verified, matrix operand order checked by reading, not run on hardware
//
#include <hip/hip_runtime.h>
#include <hip/hip_bf16.h>

typedef __attribute__((ext_vector_type(16))) _Float16 v16h;
typedef __attribute__((ext_vector_type(8)))  float    v8f;

#define NF 2000
#define NC 80
#define LGT 81
#define BATCH 2
#define NP 1000
#define HWDIM 128
#define IMG (HWDIM*HWDIM)
#define NCAND 1000
#define NSEL 100
#define NMS_THR_F 0.5f
#define OFF_SCALE 129.0f
#define SPB (NP*NC) /* 80000 scores per batch */

/* output layout (floats): labels(200) | masks(200*16384) | scores(200) | batch_ids(200) */
#define O_LAB   0
#define O_MASK  (BATCH*NSEL)
#define O_SCORE (O_MASK + BATCH*NSEL*IMG)
#define O_BID   (O_SCORE + BATCH*NSEL)

/* workspace layout (byte offsets) */
#define WS_BOXES   0        /* 2000 * float4                */
#define WS_VALID   32768    /* 2000 * int                   */
#define WS_CSCORE  40960    /* 2000 * float                 */
#define WS_CLABEL  49152    /* 2000 * int                   */
#define WS_CFEAT   57344    /* 2000 * int                   */
#define WS_CVALID  65536    /* 2000 * int                   */
#define WS_CBOX    73728    /* 2000 * float4                */
#define WS_SUP     106496   /* 2 * 32768 * u32 suppression  */
#define WS_SEL     368640   /* 200 * int                    */

/* ---------------- Kernel 1: mask -> valid + bbox, WMMA row sums ----------- */
/* grid 2000, block 256 (8 waves). Wave w owns rows [16w,16w+16).
   Column-any via __ballot (v_cmp masks are already in SGPRs) -> wave-uniform
   words, single ds_or per wave. Row-any from WMMA row sums, folded with two
   readlanes. No uniform-address atomic expansion loops.                      */
__global__ void k_maskbox(const float* __restrict__ seg,
                          float4* __restrict__ boxes, int* __restrict__ valid) {
  __shared__ unsigned rowAny[4];
  __shared__ unsigned colAny[4];
  const int f   = blockIdx.x;
  const int tid = threadIdx.x;
  if (tid < 4) { rowAny[tid] = 0u; colAny[tid] = 0u; }
  __syncthreads();

  const int w = tid >> 5, lane = tid & 31;
  const int m = lane & 15, g = lane >> 4;
  const float* rowp = seg + (size_t)f * IMG + (size_t)(16 * w + m) * HWDIM;

  v16h bones;
#pragma unroll
  for (int e = 0; e < 16; ++e) bones[e] = (_Float16)1.0f;
  v8f acc = {};
  unsigned colw[4];

#pragma unroll
  for (int kt = 0; kt < 4; ++kt) {
    const int c0 = 32 * kt + 8 * g;                  /* ISA A-layout column base */
    float4 A0 = *(const float4*)(rowp + c0);
    float4 A1 = *(const float4*)(rowp + c0 + 4);
    float4 B0 = *(const float4*)(rowp + c0 + 16);
    float4 B1 = *(const float4*)(rowp + c0 + 20);
    float va[8] = {A0.x, A0.y, A0.z, A0.w, A1.x, A1.y, A1.z, A1.w};
    float vb[8] = {B0.x, B0.y, B0.z, B0.w, B1.x, B1.y, B1.z, B1.w};
    v16h af;
    unsigned cw = 0u;
#pragma unroll
    for (int e = 0; e < 8; ++e) {
      const bool pa = va[e] > 0.0f, pb = vb[e] > 0.0f;
      af[e]     = pa ? (_Float16)1.0f : (_Float16)0.0f;
      af[8 + e] = pb ? (_Float16)1.0f : (_Float16)0.0f;
      /* ballot bit l = lane l's predicate; lanes 0-15 are the g=0 column set,
         lanes 16-31 the g=1 set -> OR over all 16 rows of this band at once */
      const unsigned balA = (unsigned)__ballot(pa);
      const unsigned balB = (unsigned)__ballot(pb);
      cw |= ((balA & 0xffffu) ? 1u : 0u) << e;
      cw |= ((balA >> 16)     ? 1u : 0u) << (8 + e);
      cw |= ((balB & 0xffffu) ? 1u : 0u) << (16 + e);
      cw |= ((balB >> 16)     ? 1u : 0u) << (24 + e);
    }
    colw[kt] = cw;                                   /* wave-uniform */
    /* row sums: D = A(16x32 indicator) x ones(32x16) + C */
    acc = __builtin_amdgcn_wmma_f32_16x16x32_f16(false, af, false, bones,
                                                 (short)0, acc, false, false);
  }

  /* D layout: lane l, elem r -> row M = r + 8*(l>=16); rows 16w+8g .. +7 here */
  unsigned rb = 0u;
#pragma unroll
  for (int r = 0; r < 8; ++r) if (acc[r] > 0.5f) rb |= 1u << r;
  const unsigned rlow  = (unsigned)__shfl((int)rb, 0);   /* rows 16w+0..7  */
  const unsigned rhigh = (unsigned)__shfl((int)rb, 16);  /* rows 16w+8..15 */
  if (lane == 0) {
    const unsigned band16 = (rlow & 0xffu) | ((rhigh & 0xffu) << 8);
    atomicOr(&rowAny[w >> 1], band16 << ((w & 1) << 4));
    atomicOr(&colAny[0], colw[0]);
    atomicOr(&colAny[1], colw[1]);
    atomicOr(&colAny[2], colw[2]);
    atomicOr(&colAny[3], colw[3]);
  }
  __syncthreads();

  if (tid == 0) {
    int y1 = -1, y2 = -1, x1 = -1, x2 = -1;
    for (int i2 = 0; i2 < 4; ++i2)  if (rowAny[i2]) { y1 = i2*32 + (__ffs((int)rowAny[i2]) - 1); break; }
    for (int i2 = 3; i2 >= 0; --i2) if (rowAny[i2]) { y2 = i2*32 + (31 - __clz((int)rowAny[i2])); break; }
    for (int i2 = 0; i2 < 4; ++i2)  if (colAny[i2]) { x1 = i2*32 + (__ffs((int)colAny[i2]) - 1); break; }
    for (int i2 = 3; i2 >= 0; --i2) if (colAny[i2]) { x2 = i2*32 + (31 - __clz((int)colAny[i2])); break; }
    const bool vld = (y1 >= 0);
    boxes[f] = vld ? make_float4((float)x1, (float)y1, (float)x2, (float)y2)
                   : make_float4(1e9f, 1e9f, -1e9f, -1e9f);
    valid[f] = vld ? 1 : 0;
  }
}

/* -------- Kernel 2: sigmoid scores, radix-select + bitonic top-1000 ------- */
/* grid 2 (one block per batch), block 1024.                                  */
__global__ void k_topk(const float* __restrict__ cls, const int* __restrict__ valid,
                       const float4* __restrict__ boxes,
                       float* __restrict__ cscore, int* __restrict__ clabel,
                       int* __restrict__ cfeat, int* __restrict__ cvalid,
                       float4* __restrict__ cbox) {
  __shared__ unsigned hist[256];
  __shared__ unsigned long long buf[2048];
  __shared__ unsigned sc[8]; /* 0:b1 1:b2 2:b3 3:counter 4:prefixA 5:prefixB */
  const int b = blockIdx.x, tid = threadIdx.x;

  auto key_of = [&](int j) -> unsigned {
    const int feat = j / NC, c = j - feat * NC;
    const int gfi  = b * NP + feat;
    if (!valid[gfi]) return 0u;
    const float x = cls[(size_t)gfi * LGT + c];
    const float s = 1.0f / (1.0f + __expf(-x));  /* s in (0,1): bits order = value order */
    return __float_as_uint(s);
  };

  /* pass A: byte[31:24] */
  if (tid < 256) hist[tid] = 0u;
  __syncthreads();
  for (int j = tid; j < SPB; j += 1024) { unsigned k = key_of(j); if (k) atomicAdd(&hist[k >> 24], 1u); }
  __syncthreads();
  if (tid == 0) {
    unsigned cum = 0u, b1 = 0u;
    for (int x = 255; x >= 0; --x) { if (cum + hist[x] >= NCAND) { b1 = (unsigned)x; break; } cum += hist[x]; }
    sc[0] = b1; sc[4] = cum;
  }
  __syncthreads();
  /* pass B: byte[23:16] within b1 */
  if (tid < 256) hist[tid] = 0u;
  __syncthreads();
  { const unsigned b1 = sc[0];
    for (int j = tid; j < SPB; j += 1024) { unsigned k = key_of(j); if (k && (k >> 24) == b1) atomicAdd(&hist[(k >> 16) & 255u], 1u); } }
  __syncthreads();
  if (tid == 0) {
    unsigned cum = sc[4], b2 = 0u;
    for (int x = 255; x >= 0; --x) { if (cum + hist[x] >= NCAND) { b2 = (unsigned)x; break; } cum += hist[x]; }
    sc[1] = b2; sc[5] = cum;
  }
  __syncthreads();
  /* pass C: byte[15:8] within (b1,b2) */
  if (tid < 256) hist[tid] = 0u;
  __syncthreads();
  { const unsigned hi16 = (sc[0] << 8) | sc[1];
    for (int j = tid; j < SPB; j += 1024) { unsigned k = key_of(j); if (k && (k >> 16) == hi16) atomicAdd(&hist[(k >> 8) & 255u], 1u); } }
  __syncthreads();
  if (tid == 0) {
    unsigned cum = sc[5], b3 = 0u;
    for (int x = 255; x >= 0; --x) { if (cum + hist[x] >= NCAND) { b3 = (unsigned)x; break; } cum += hist[x]; }
    sc[2] = b3; sc[3] = 0u;
  }
  __syncthreads();
  /* pass D: compact keys >= T, composite (key<<32)|~j -> unique, stable ties */
  const unsigned T = (sc[0] << 24) | (sc[1] << 16) | (sc[2] << 8);
  for (int j = tid; j < SPB; j += 1024) {
    unsigned k = key_of(j);
    if (k && k >= T) {
      unsigned pos = atomicAdd(&sc[3], 1u);
      if (pos < 2048u)
        buf[pos] = ((unsigned long long)k << 32) | (unsigned)(~(unsigned)j);
    }
  }
  __syncthreads();
  { const unsigned cnt = sc[3] < 2048u ? sc[3] : 2048u;
    for (unsigned i = cnt + tid; i < 2048u; i += 1024u) buf[i] = 0ull; }
  /* bitonic sort, descending */
  for (unsigned kk = 2; kk <= 2048u; kk <<= 1) {
    for (unsigned jj = kk >> 1; jj > 0; jj >>= 1) {
      __syncthreads();
      for (unsigned i = tid; i < 2048u; i += 1024u) {
        const unsigned l = i ^ jj;
        if (l > i) {
          const unsigned long long a = buf[i], c = buf[l];
          const bool up = ((i & kk) == 0u);
          if (up ? (a < c) : (a > c)) { buf[i] = c; buf[l] = a; }
        }
      }
    }
  }
  __syncthreads();
  if (tid < NCAND) {
    const unsigned long long e = buf[tid];
    const unsigned key = (unsigned)(e >> 32);
    const int o = b * NCAND + tid;
    if (key) {
      const unsigned j = ~(unsigned)(e & 0xffffffffu);
      const int feat = (int)(j / NC), lab = (int)(j % NC);
      cscore[o] = __uint_as_float(key);
      clabel[o] = lab; cfeat[o] = feat; cvalid[o] = 1;
      const float4 bx = boxes[b * NP + feat];
      const float off = OFF_SCALE * (float)lab;
      cbox[o] = make_float4(bx.x + off, bx.y + off, bx.z + off, bx.w + off);
    } else {
      cscore[o] = -__builtin_inff();
      clabel[o] = 0; cfeat[o] = 0; cvalid[o] = 0;
      cbox[o] = make_float4(1e9f, 1e9f, -1e9f, -1e9f);
    }
  }
}

/* ------- Kernel 3: pairwise IoU bitmask + serial NMS + selection ---------- */
/* grid 2, block 1024.                                                        */
__global__ void k_nms(const float4* __restrict__ cbox, const int* __restrict__ cvalid,
                      unsigned* __restrict__ sup, int* __restrict__ sel) {
  __shared__ float4  bxl[NCAND];
  __shared__ float   areal[NCAND];
  __shared__ unsigned keepw[32];
  __shared__ int     sb[2][1024];
  const int b = blockIdx.x, tid = threadIdx.x;

  if (tid < 32) keepw[tid] = 0u;
  __syncthreads();
  if (tid < NCAND) {
    const float4 v = cbox[b * NCAND + tid];
    bxl[tid] = v;
    areal[tid] = fmaxf(v.z - v.x, 0.0f) * fmaxf(v.w - v.y, 0.0f);
    if (cvalid[b * NCAND + tid]) atomicOr(&keepw[tid >> 5], 1u << (tid & 31));
  }
  __syncthreads();

  unsigned* supB = sup + (size_t)b * 32768;
  for (int w0 = tid; w0 < NCAND * 32; w0 += 1024) {
    const int i = w0 >> 5, wj = w0 & 31;
    const float4 bi = bxl[i];
    const float  ai = areal[i];
    unsigned bits = 0u;
    for (int t2 = 0; t2 < 32; ++t2) {
      const int j = (wj << 5) | t2;
      if (j > i && j < NCAND) {
        const float4 bj = bxl[j];
        const float ix1 = fmaxf(bi.x, bj.x), iy1 = fmaxf(bi.y, bj.y);
        const float ix2 = fminf(bi.z, bj.z), iy2 = fminf(bi.w, bj.w);
        const float inter = fmaxf(ix2 - ix1, 0.0f) * fmaxf(iy2 - iy1, 0.0f);
        const float uni = ai + areal[j] - inter;
        const float iou = (uni > 0.0f) ? (inter / uni) : 0.0f;
        if (iou > NMS_THR_F) bits |= 1u << t2;
      }
    }
    supB[w0] = bits;
  }
  __syncthreads();

  /* serial keep scan on one wave32: lane k owns keep word k.
     Double-buffered 32-row chunks: one batched load latency per 32 rows. */
  if (tid < 32) {
    unsigned keepK = keepw[tid];
    unsigned cur[32], nxt[32];
#pragma unroll
    for (int r = 0; r < 32; ++r) cur[r] = supB[r * 32 + tid];
    for (int c = 0; c < 32; ++c) {
      const int cn = c + 1;
#pragma unroll
      for (int r = 0; r < 32; ++r) {
        const int i = cn * 32 + r;
        nxt[r] = (i < NCAND) ? supB[i * 32 + tid] : 0u;
      }
#pragma unroll
      for (int r = 0; r < 32; ++r) {
        const int i = c * 32 + r;
        if (i < NCAND) {
          const unsigned kw = __shfl(keepK, c); /* keep word for rows of chunk c */
          if ((kw >> r) & 1u) keepK &= ~cur[r];
        }
      }
#pragma unroll
      for (int r = 0; r < 32; ++r) cur[r] = nxt[r];
    }
    keepw[tid] = keepK;
  }
  __syncthreads();

  /* selection order: kept (score-desc = index order) then not-kept (index order) */
  int kflag = 0;
  if (tid < NCAND) kflag = (int)((keepw[tid >> 5] >> (tid & 31)) & 1u);
  sb[0][tid] = kflag;
  int src = 0;
  for (int off = 1; off < 1024; off <<= 1) {
    __syncthreads();
    int v = sb[src][tid];
    if (tid >= off) v += sb[src][tid - off];
    sb[1 - src][tid] = v;
    src ^= 1;
  }
  __syncthreads();
  if (tid < NCAND) {
    const int incl = sb[src][tid];
    const int K    = sb[src][NCAND - 1];
    const int excl = incl - kflag;
    const int pos  = kflag ? excl : (K + (tid - excl));
    if (pos < NSEL) sel[b * NSEL + pos] = tid;
  }
}

/* --------------- Kernel 4a: labels / scores / batch ids ------------------- */
__global__ void k_final(const int* __restrict__ sel, const float* __restrict__ cscore,
                        const int* __restrict__ clabel, float* __restrict__ out) {
  const int s = threadIdx.x;
  if (s >= BATCH * NSEL) return;
  const int b = s / NSEL;
  const int gi = b * NCAND + sel[s];
  out[O_LAB + s]   = (float)clabel[gi];
  out[O_SCORE + s] = cscore[gi];
  out[O_BID + s]   = (float)b;
}

/* --------------- Kernel 4b: emit 200 selected masks ----------------------- */
/* grid 200, block 256; recompute seg>0 only for winners (13 MB not 131 MB). */
__global__ void k_masks(const float* __restrict__ seg, const int* __restrict__ sel,
                        const int* __restrict__ cfeat, float* __restrict__ out) {
  const int s = blockIdx.x;
  const int b = s / NSEL;
  const int feat = cfeat[b * NCAND + sel[s]];
  const float* src = seg + (size_t)(b * NP + feat) * IMG;
  float* dst = out + O_MASK + (size_t)s * IMG;
  for (int e = threadIdx.x * 4; e < IMG; e += 256 * 4) {
    if (e + 1024 < IMG) __builtin_prefetch(src + e + 1024, 0, 0);
    const float4 v = *(const float4*)(src + e);
    float4 o;
    o.x = v.x > 0.0f ? 1.0f : 0.0f;
    o.y = v.y > 0.0f ? 1.0f : 0.0f;
    o.z = v.z > 0.0f ? 1.0f : 0.0f;
    o.w = v.w > 0.0f ? 1.0f : 0.0f;
    *(float4*)(dst + e) = o;
  }
}

extern "C" void kernel_launch(void* const* d_in, const int* in_sizes, int n_in,
                              void* d_out, int out_size, void* d_ws, size_t ws_size,
                              hipStream_t stream) {
  const float* cls = (const float*)d_in[0]; /* (2000, 81)        */
  const float* seg = (const float*)d_in[1]; /* (2000, 128, 128)  */
  float* out = (float*)d_out;
  char*  w   = (char*)d_ws;

  float4*   boxes  = (float4*)(w + WS_BOXES);
  int*      valid  = (int*)(w + WS_VALID);
  float*    cscore = (float*)(w + WS_CSCORE);
  int*      clabel = (int*)(w + WS_CLABEL);
  int*      cfeat  = (int*)(w + WS_CFEAT);
  int*      cvalid = (int*)(w + WS_CVALID);
  float4*   cbox   = (float4*)(w + WS_CBOX);
  unsigned* sup    = (unsigned*)(w + WS_SUP);
  int*      sel    = (int*)(w + WS_SEL);

  k_maskbox<<<NF, 256, 0, stream>>>(seg, boxes, valid);
  k_topk<<<BATCH, 1024, 0, stream>>>(cls, valid, boxes, cscore, clabel, cfeat, cvalid, cbox);
  k_nms<<<BATCH, 1024, 0, stream>>>(cbox, cvalid, sup, sel);
  k_final<<<1, 256, 0, stream>>>(sel, cscore, clabel, out);
  k_masks<<<BATCH * NSEL, 256, 0, stream>>>(seg, sel, cfeat, out);
}
